// SwinTransformerBlock_36197984371056
// MI455X (gfx1250) — matrix operand order repaired and checked by reading, not compiled, not run on hardware
//
#include <hip/hip_runtime.h>
#include <hip/hip_bf16.h>

typedef _Float16 half_t;
typedef half_t v8h  __attribute__((ext_vector_type(8)));
typedef half_t v16h __attribute__((ext_vector_type(16)));
typedef float  v8f  __attribute__((ext_vector_type(8)));

#define TOK      50176      // B * 56 * 56 = 16*3136
#define CDIM     384
#define NHEAD    12
#define NWIN_T   1024       // B * NW = 16 * 64
#define HIDDIM   1536
#define QKVDIM   1152
#define SCALE_Q  0.1767766952966369f   // 32^-0.5

// windowed-token index -> original pixel index (shift by +3 roll, window reverse)
__device__ __forceinline__ int map_w2p(int t) {
  int b   = t / 3136;
  int rr  = t - b * 3136;
  int wid = rr / 49;
  int n   = rr - wid * 49;
  int sh  = (wid >> 3) * 7 + n / 7;
  int sw  = (wid & 7) * 7 + n % 7;
  int oh  = sh + 3; if (oh >= 56) oh -= 56;
  int ow  = sw + 3; if (ow >= 56) ow -= 56;
  return b * 3136 + oh * 56 + ow;
}

__device__ __forceinline__ int seg56(int x) { return (x < 49) ? 0 : ((x < 53) ? 1 : 2); }

// ---------------- f32 -> f16 weight convert ----------------
__global__ __launch_bounds__(256) void cvt_f32_f16(const float* __restrict__ s,
                                                   half_t* __restrict__ d, int n) {
  int i = blockIdx.x * 256 + threadIdx.x;
  if (i < n) d[i] = (half_t)s[i];
}

// ---------------- LayerNorm (+optional gather permutation) ----------------
// out row = blockIdx.x ; source row = permute ? map_w2p(row) : row
__global__ __launch_bounds__(128) void ln_kernel(const float* __restrict__ X,
                                                 const float* __restrict__ g,
                                                 const float* __restrict__ bt,
                                                 half_t* __restrict__ out, int permute) {
  __shared__ float red[128];
  const int tid = threadIdx.x;
  const int row = blockIdx.x;
  const int srow = permute ? map_w2p(row) : row;
  const float* xp = X + srow * CDIM;
  float x0 = xp[tid], x1 = xp[tid + 128], x2 = xp[tid + 256];
  red[tid] = x0 + x1 + x2;
  __syncthreads();
  for (int off = 64; off > 0; off >>= 1) { if (tid < off) red[tid] += red[tid + off]; __syncthreads(); }
  const float mean = red[0] * (1.0f / 384.0f);
  __syncthreads();
  float d0 = x0 - mean, d1 = x1 - mean, d2 = x2 - mean;
  red[tid] = d0 * d0 + d1 * d1 + d2 * d2;
  __syncthreads();
  for (int off = 64; off > 0; off >>= 1) { if (tid < off) red[tid] += red[tid + off]; __syncthreads(); }
  const float rstd = rsqrtf(red[0] * (1.0f / 384.0f) + 1e-5f);
  half_t* op = out + row * CDIM;
  op[tid]       = (half_t)(d0 * rstd * g[tid]       + bt[tid]);
  op[tid + 128] = (half_t)(d1 * rstd * g[tid + 128] + bt[tid + 128]);
  op[tid + 256] = (half_t)(d2 * rstd * g[tid + 256] + bt[tid + 256]);
}

// ---------------- WMMA GEMM: C[M,NC] = A[M,KC] @ W[KC,NC], fused epilogues ----
// block = 256 threads = 8 waves arranged 4(M) x 2(N); WG tile 64M x 128N.
// Double-buffered W tile in LDS (ping-pong): one barrier per K-step, global
// fetch of tile k+1 overlaps WMMAs of tile k.
// EPI 0: QKV split/scale -> o0/o1/o2(f16)   EPI 1: proj + scatter + residual -> of
// EPI 2: +bias, exact GELU -> o0(f16)       EPI 3: of += gemm + bias
template<int KC, int NC, int EPI>
__global__ __launch_bounds__(256) void gemm_wmma(
    const half_t* __restrict__ A, const half_t* __restrict__ W,
    const float* __restrict__ bias,
    half_t* __restrict__ o0, half_t* __restrict__ o1, half_t* __restrict__ o2,
    float* __restrict__ of, const float* __restrict__ xres) {
  __shared__ __attribute__((aligned(64))) half_t WB[2][8 * 32 * 16];  // B-frag order
  const int tid  = threadIdx.x;
  const int lane = tid & 31;
  const int wave = tid >> 5;
  const int wm   = wave & 3;
  const int wn   = wave >> 2;
  const int l15  = lane & 15;
  const int hiH  = lane >> 4;
  const int Mbase = blockIdx.y * 64;
  const int Nbase = blockIdx.x * 128;

  v8f acc[4] = {};

  const int arow = Mbase + wm * 16 + l15;
  const half_t* __restrict__ Ar = A + arow * KC;
  const int k0 = hiH * 16;          // contiguous A-layout: lanes 16-31 start at K=16
  const int wk  = tid >> 3;         // 0..31  (K row of W tile)
  const int wng = tid & 7;          // 0..7   (16-col group)
  const int lp  = (wk >> 4) << 4;   // lane-group (K half)
  const int ii  = wk & 15;          // element index inside frag

  // prologue: stage tile 0
  {
    const half_t* wp = W + wk * NC + Nbase + wng * 16;
    v8h w0 = *(const v8h*)(wp);
    v8h w1 = *(const v8h*)(wp + 8);
    #pragma unroll
    for (int j = 0; j < 8; ++j) {
      WB[0][(wng * 32 + lp + j) * 16 + ii]     = w0[j];
      WB[0][(wng * 32 + lp + 8 + j) * 16 + ii] = w1[j];
    }
  }

  int cur = 0;
  for (int kk = 0; kk < KC; kk += 32) {
    __syncthreads();   // WB[cur] ready for all waves; WB[cur^1] free

    const bool more = (kk + 32 < KC);
    v8h nw0 = {}, nw1 = {};
    if (more) {        // issue global fetch of next W tile before the WMMAs
      const half_t* wp = W + (kk + 32 + wk) * NC + Nbase + wng * 16;
      nw0 = *(const v8h*)(wp);
      nw1 = *(const v8h*)(wp + 8);
      __builtin_prefetch((const void*)(Ar + kk + 32), 0, 0);
    }

    v8h alo = *(const v8h*)(Ar + kk + k0);        // K = k0..k0+7
    v8h ahi = *(const v8h*)(Ar + kk + k0 + 8);    // K = k0+8..k0+15
    v16h af = __builtin_shufflevector(alo, ahi, 0,1,2,3,4,5,6,7,8,9,10,11,12,13,14,15);

    #pragma unroll
    for (int nt = 0; nt < 4; ++nt) {
      const int ntg = wn * 4 + nt;
      v16h bf = *(const v16h*)(&WB[cur][(ntg * 32 + lane) * 16]);
      acc[nt] = __builtin_amdgcn_wmma_f32_16x16x32_f16(
          false, af, false, bf, (short)0, acc[nt], false, false);
    }

    if (more) {        // store prefetched tile into the other buffer
      #pragma unroll
      for (int j = 0; j < 8; ++j) {
        WB[cur ^ 1][(wng * 32 + lp + j) * 16 + ii]     = nw0[j];
        WB[cur ^ 1][(wng * 32 + lp + 8 + j) * 16 + ii] = nw1[j];
      }
    }
    cur ^= 1;
  }

  #pragma unroll
  for (int r = 0; r < 8; ++r) {
    const int grow = Mbase + wm * 16 + (hiH ? r + 8 : r);
    int prow = 0;
    if constexpr (EPI == 1) prow = map_w2p(grow);
    #pragma unroll
    for (int nt = 0; nt < 4; ++nt) {
      const int gcol = Nbase + wn * 64 + nt * 16 + l15;
      float v = acc[nt][r] + bias[gcol];
      if constexpr (EPI == 0) {
        const int qi = gcol / 384;
        const int hd = gcol - qi * 384;     // head*32 + d
        if (qi == 0)      o0[grow * 384 + hd] = (half_t)(v * SCALE_Q);
        else if (qi == 1) o1[grow * 384 + hd] = (half_t)v;
        else              o2[grow * 384 + hd] = (half_t)v;
      } else if constexpr (EPI == 1) {
        of[prow * 384 + gcol] = xres[prow * 384 + gcol] + v;
      } else if constexpr (EPI == 2) {
        float gg = 0.5f * v * (1.0f + erff(v * 0.70710678f));
        o0[grow * HIDDIM + gcol] = (half_t)gg;
      } else {
        of[grow * 384 + gcol] += v;
      }
    }
  }
}

// ---------------- windowed attention: one block per (window, head) ----------
__global__ __launch_bounds__(128) void attn_wmma(
    const half_t* __restrict__ Qb, const half_t* __restrict__ Kb,
    const half_t* __restrict__ Vb, const float* __restrict__ btab,
    half_t* __restrict__ O) {
  __shared__ __attribute__((aligned(64))) half_t Qs[64 * 32];
  __shared__ __attribute__((aligned(64))) half_t Ks[64 * 32];
  __shared__ __attribute__((aligned(64))) half_t Vt[32 * 64];     // transposed V
  __shared__ __attribute__((aligned(64))) float  Sb[4 * 16 * 64];
  __shared__ __attribute__((aligned(64))) half_t Pb[4 * 16 * 64];

  const int h    = blockIdx.x;          // head 0..11
  const int win  = blockIdx.y;          // 0..1023 (b*64 + wid)
  const int wid  = win & 63;
  const int tid  = threadIdx.x;
  const int lane = tid & 31;
  const int wave = tid >> 5;
  const int l15  = lane & 15;
  const int hiH  = lane >> 4;

  {   // stage Q, K (row-major) and V (transposed), zero-pad rows 49..63
    const int row = tid >> 1;
    const int off = (tid & 1) << 4;
    v8h q0 = {}, q1 = {}, ka = {}, kb2 = {}, va = {}, vb2 = {};
    if (row < 49) {
      const int gb = (win * 49 + row) * 384 + h * 32 + off;
      q0  = *(const v8h*)(Qb + gb);  q1  = *(const v8h*)(Qb + gb + 8);
      ka  = *(const v8h*)(Kb + gb);  kb2 = *(const v8h*)(Kb + gb + 8);
      va  = *(const v8h*)(Vb + gb);  vb2 = *(const v8h*)(Vb + gb + 8);
    }
    #pragma unroll
    for (int j = 0; j < 8; ++j) {
      Qs[row * 32 + off + j]       = q0[j];
      Qs[row * 32 + off + 8 + j]   = q1[j];
      Ks[row * 32 + off + j]       = ka[j];
      Ks[row * 32 + off + 8 + j]   = kb2[j];
      Vt[(off + j) * 64 + row]     = va[j];
      Vt[(off + 8 + j) * 64 + row] = vb2[j];
    }
  }
  __syncthreads();

  {   // S = Q K^T  (head_dim=32 -> one WMMA per 16x16 tile)
    const int qr = wave * 16 + l15;
    const int k0 = hiH * 16;
    v8h alo = *(const v8h*)(&Qs[qr * 32 + k0]);
    v8h ahi = *(const v8h*)(&Qs[qr * 32 + k0 + 8]);
    v16h aq = __builtin_shufflevector(alo, ahi, 0,1,2,3,4,5,6,7,8,9,10,11,12,13,14,15);
    #pragma unroll
    for (int nt = 0; nt < 4; ++nt) {
      const int kc = nt * 16 + l15;
      v16h bk = *(const v16h*)(&Ks[kc * 32 + hiH * 16]);
      v8f cz = {};
      v8f s = __builtin_amdgcn_wmma_f32_16x16x32_f16(false, aq, false, bk, (short)0, cz, false, false);
      #pragma unroll
      for (int r = 0; r < 8; ++r)
        Sb[(wave * 16 + (hiH ? r + 8 : r)) * 64 + nt * 16 + l15] = s[r];
    }
  }
  __syncthreads();

  // rel-pos bias + shift mask + softmax; one lane per row
  if (lane < 16) {
    const int qtok = wave * 16 + lane;
    float*  Srow = &Sb[(wave * 16 + lane) * 64];
    half_t* Prow = &Pb[(wave * 16 + lane) * 64];
    if (qtok < 49) {
      const int wr = (wid >> 3) * 7, wc = (wid & 7) * 7;
      const int r1 = qtok / 7, c1 = qtok - r1 * 7;
      const int ridq = seg56(wr + r1) * 3 + seg56(wc + c1);
      float mx = -1e30f;
      for (int j = 0; j < 49; ++j) {
        const int r2 = j / 7, c2 = j - r2 * 7;
        const int idx = (r1 - r2 + 6) * 13 + (c1 - c2 + 6);
        float sv = Srow[j] + btab[idx * NHEAD + h];
        const int ridk = seg56(wr + r2) * 3 + seg56(wc + c2);
        if (ridq != ridk) sv -= 100.0f;
        Srow[j] = sv;
        mx = fmaxf(mx, sv);
      }
      float sum = 0.0f;
      for (int j = 0; j < 49; ++j) { float e = __expf(Srow[j] - mx); sum += e; Srow[j] = e; }
      const float inv = 1.0f / sum;
      for (int j = 0; j < 49; ++j) Prow[j] = (half_t)(Srow[j] * inv);
      for (int j = 49; j < 64; ++j) Prow[j] = (half_t)0.0f;
    } else {
      for (int j = 0; j < 64; ++j) Prow[j] = (half_t)0.0f;
    }
  }
  __syncthreads();

  // O = P @ V : K = 64 (two WMMA k-steps), N = 32 (two 16-col tiles)
  v8f accO[2] = {};
  const int pr = wave * 16 + l15;
  const int k0 = hiH * 16;
  #pragma unroll
  for (int ks = 0; ks < 2; ++ks) {
    v8h plo = *(const v8h*)(&Pb[pr * 64 + ks * 32 + k0]);
    v8h phi = *(const v8h*)(&Pb[pr * 64 + ks * 32 + k0 + 8]);
    v16h ap = __builtin_shufflevector(plo, phi, 0,1,2,3,4,5,6,7,8,9,10,11,12,13,14,15);
    #pragma unroll
    for (int nt = 0; nt < 2; ++nt) {
      v16h bv = *(const v16h*)(&Vt[(nt * 16 + l15) * 64 + ks * 32 + hiH * 16]);
      accO[nt] = __builtin_amdgcn_wmma_f32_16x16x32_f16(
          false, ap, false, bv, (short)0, accO[nt], false, false);
    }
  }
  #pragma unroll
  for (int nt = 0; nt < 2; ++nt) {
    #pragma unroll
    for (int r = 0; r < 8; ++r) {
      const int tok = wave * 16 + (hiH ? r + 8 : r);
      if (tok < 49)
        O[(win * 49 + tok) * 384 + h * 32 + nt * 16 + l15] = (half_t)accO[nt][r];
    }
  }
}

extern "C" void kernel_launch(void* const* d_in, const int* in_sizes, int n_in,
                              void* d_out, int out_size, void* d_ws, size_t ws_size,
                              hipStream_t stream) {
  (void)in_sizes; (void)n_in; (void)out_size; (void)ws_size;
  const float* x      = (const float*)d_in[0];
  const float* qkv_w  = (const float*)d_in[1];
  const float* qkv_b  = (const float*)d_in[2];
  const float* proj_w = (const float*)d_in[3];
  const float* proj_b = (const float*)d_in[4];
  const float* btab   = (const float*)d_in[5];
  const float* ln1_g  = (const float*)d_in[6];
  const float* ln1_b  = (const float*)d_in[7];
  const float* ln2_g  = (const float*)d_in[8];
  const float* ln2_b  = (const float*)d_in[9];
  const float* mlp_w1 = (const float*)d_in[10];
  const float* mlp_b1 = (const float*)d_in[11];
  const float* mlp_w2 = (const float*)d_in[12];
  const float* mlp_b2 = (const float*)d_in[13];
  float* out = (float*)d_out;

  const size_t TC = (size_t)TOK * CDIM;       // 19,267,584 halfs
  half_t* ws    = (half_t*)d_ws;
  half_t* wqkv  = ws;                         // 384*1152
  half_t* wproj = wqkv  + 442368;             // 384*384
  half_t* wm1   = wproj + 147456;             // 384*1536
  half_t* wm2   = wm1   + 589824;             // 1536*384
  half_t* h2    = wm2   + 589824;             // [TOK, C] f16 (LN2 out)
  half_t* hwin  = h2   + TC;                  // [TOK, C] windowed LN1 tokens
  half_t* qb    = hwin + TC;
  half_t* kb    = hwin + 2 * TC;
  half_t* vb    = hwin + 3 * TC;
  half_t* attn  = hwin;                       // alias: hwin dead after QKV GEMM
  half_t* hid   = hwin;                       // alias: spans hwin..vb = TOK*1536

  cvt_f32_f16<<<(442368 + 255) / 256, 256, 0, stream>>>(qkv_w,  wqkv, 442368);
  cvt_f32_f16<<<(147456 + 255) / 256, 256, 0, stream>>>(proj_w, wproj, 147456);
  cvt_f32_f16<<<(589824 + 255) / 256, 256, 0, stream>>>(mlp_w1, wm1, 589824);
  cvt_f32_f16<<<(589824 + 255) / 256, 256, 0, stream>>>(mlp_w2, wm2, 589824);

  // LN1 + shift + window partition (gather), f16
  ln_kernel<<<TOK, 128, 0, stream>>>(x, ln1_g, ln1_b, hwin, 1);
  // QKV projection
  gemm_wmma<384, QKVDIM, 0><<<dim3(QKVDIM / 128, TOK / 64), 256, 0, stream>>>(
      hwin, wqkv, qkv_b, qb, kb, vb, nullptr, nullptr);
  // windowed attention
  attn_wmma<<<dim3(NHEAD, NWIN_T), 128, 0, stream>>>(qb, kb, vb, btab, attn);
  // output projection + window reverse + un-shift + residual -> d_out (= x1)
  gemm_wmma<384, 384, 1><<<dim3(384 / 128, TOK / 64), 256, 0, stream>>>(
      attn, wproj, proj_b, nullptr, nullptr, nullptr, out, x);
  // LN2
  ln_kernel<<<TOK, 128, 0, stream>>>(out, ln2_g, ln2_b, h2, 0);
  // MLP fc1 + exact GELU
  gemm_wmma<384, HIDDIM, 2><<<dim3(HIDDIM / 128, TOK / 64), 256, 0, stream>>>(
      h2, wm1, mlp_b1, hid, nullptr, nullptr, nullptr, nullptr);
  // MLP fc2, accumulate into x1 -> final output
  gemm_wmma<HIDDIM, 384, 3><<<dim3(384 / 128, TOK / 64), 256, 0, stream>>>(
      hid, wm2, mlp_b2, nullptr, nullptr, nullptr, out, nullptr);
}
